// SA_Block_PoolKV_38130719654098
// MI455X (gfx1250) — matrix operand order repaired and checked by reading, not compiled, not run on hardware
//
#include <hip/hip_runtime.h>
#include <hip/hip_bf16.h>

typedef __attribute__((ext_vector_type(16))) _Float16 v16h;
typedef __attribute__((ext_vector_type(8)))  float    v8f;

#define CDIM  64
#define NPIX  16384     // H*W
#define MKEYS 4096      // pooled keys
#define TQ    64        // queries per block
#define TK    64        // keys per chunk
#define NCH   (MKEYS / TK)
#define QP    72        // f16 LDS pitch (144 B rows -> 16B aligned)
#define SP    68        // f32 LDS pitch (pad)

// ---------------------------------------------------------------------------
// Kernel 1: 1x1 conv (+ fused 2x2 avg-pool for k/v, since pool(Wx+b)=W pool(x)+b)
// mode 0: full-res, out[b][n][c]  (query-major, f16, WMMA A-side)
// mode 1: pooled,  out[b][c][m]  (channel-major, f16, WMMA B-side)
// ---------------------------------------------------------------------------
__global__ __launch_bounds__(256) void qkv_kernel(
    const float* __restrict__ x, const float* __restrict__ W,
    const float* __restrict__ bias, _Float16* __restrict__ out, int mode)
{
  __shared__ float Wl[CDIM * CDIM];
  __shared__ float bl[CDIM];
  const int tid = threadIdx.x;
  for (int i = tid; i < CDIM * CDIM; i += 256) Wl[i] = W[i];
  if (tid < CDIM) bl[tid] = bias[tid];
  __syncthreads();

  const int  npix = (mode == 0) ? NPIX : MKEYS;
  const long gp   = (long)blockIdx.x * 256 + tid;
  const int  b    = (int)(gp / npix);
  const int  p    = (int)(gp % npix);

  float xr[CDIM];
  if (mode == 0) {
    const float* xb = x + (long)b * CDIM * NPIX + p;
#pragma unroll
    for (int c = 0; c < CDIM; ++c) xr[c] = xb[(long)c * NPIX];
  } else {
    const int hi = p >> 6, wi = p & 63;              // pooled coords on 64x64 grid
    const long base = (long)b * CDIM * NPIX + (long)(2 * hi) * 128 + 2 * wi;
#pragma unroll
    for (int c = 0; c < CDIM; ++c) {
      const float* xc = x + base + (long)c * NPIX;
      xr[c] = 0.25f * (xc[0] + xc[1] + xc[128] + xc[129]);
    }
  }

  for (int o = 0; o < CDIM; ++o) {
    float acc = bl[o];
#pragma unroll
    for (int c = 0; c < CDIM; ++c) acc += xr[c] * Wl[o * CDIM + c];
    if (mode == 0) out[((long)b * NPIX + p) * CDIM + o] = (_Float16)acc;
    else           out[((long)b * CDIM + o) * MKEYS + p] = (_Float16)acc;
  }
}

// ---------------------------------------------------------------------------
// Kernel 2: fused flash attention. Grid (N/64, B), 256 threads = 8 wave32.
// - Q A-fragments cached in registers for the whole kernel
// - K/V double-buffered in LDS, pre-swizzled into WMMA B-fragment layout:
//     KsB[buf][j][s][hi][nn][e] = K[c = s*32+hi*16+e][key = j*16+nn]
//     VsB[buf][j][s][hi][nn][e] = V[c = j*16+nn     ][key = s*32+hi*16+e]
//   so every lane's B-fragment is one contiguous 32-byte v16h load.
// - next chunk's global reads issue at loop top, commit to alt buffer at loop end
// ---------------------------------------------------------------------------
__global__ __launch_bounds__(256) void attn_kernel(
    const _Float16* __restrict__ qg, const _Float16* __restrict__ kg,
    const _Float16* __restrict__ vg, const float* __restrict__ xg,
    const float* __restrict__ gamma, float* __restrict__ outg)
{
  __shared__ _Float16 QPs[TQ][QP];    // Q tile, reused as P tile after frag caching
  __shared__ __attribute__((aligned(32))) _Float16 KsB[2][4][2][2][16][16];
  __shared__ __attribute__((aligned(32))) _Float16 VsB[2][4][2][2][16][16];
  __shared__ float    Ss[TQ][SP];     // score tile / final O staging
  __shared__ float rowM[TQ], rowL[TQ], rowScale[TQ];
  __shared__ float pmax[4][TQ], psum[4][TQ];

  const int tid  = threadIdx.x;
  const int wave = tid >> 5, lane = tid & 31;
  const int b     = blockIdx.y;
  const int qbase = blockIdx.x * TQ;
  const int ti = wave & 3;            // query-tile index (0..3)
  const int j0 = (wave >> 2) * 2;     // column-tile pair base (0 or 2)
  const int rr = lane & 15, hi = lane >> 4;

  // --- load Q tile (32B-aligned vector copies) ---
  {
    const int r = tid >> 2, coff = (tid & 3) * 16;
    const _Float16* src = qg + ((long)(b * NPIX + qbase + r)) * CDIM + coff;
    const v16h qv = *(const v16h*)src;
#pragma unroll
    for (int u = 0; u < 16; ++u) QPs[r][coff + u] = qv[u];
  }
  if (tid < TQ) { rowM[tid] = -1e30f; rowL[tid] = 0.f; }

  // --- per-thread K/V streaming addresses & swizzle coords ---
  const int kc = tid >> 2, koff = (tid & 3) * 16;
  const _Float16* skp = kg + (long)b * CDIM * MKEYS + (long)kc * MKEYS + koff;
  const _Float16* svp = vg + (long)b * CDIM * MKEYS + (long)kc * MKEYS + koff;
  const int ks = kc >> 5, kh = (kc >> 4) & 1, ke = kc & 15, kj = koff >> 4;
  const int vj = kc >> 4, vs = koff >> 5, vh = (koff >> 4) & 1, vn = kc & 15;

  // --- prologue: chunk 0 into buffer 0 ---
  v16h kvreg = *(const v16h*)skp;
  v16h vvreg = *(const v16h*)svp;
#pragma unroll
  for (int u = 0; u < 16; ++u) KsB[0][kj][ks][kh][u][ke] = kvreg[u];
  *(v16h*)&VsB[0][vj][vs][vh][vn][0] = vvreg;
  __syncthreads();

  // --- cache Q A-fragments in registers (Q LDS tile is dead afterwards) ---
  v16h aq0, aq1;
  {
    const int row = ti * 16 + rr;
#pragma unroll
    for (int e = 0; e < 8; ++e) {     // A-layout: K = {cb..cb+15} contiguous, cb = hi*16
      aq0[e]     = QPs[row][hi * 16 + e];
      aq0[8 + e] = QPs[row][hi * 16 + 8 + e];
      aq1[e]     = QPs[row][32 + hi * 16 + e];
      aq1[8 + e] = QPs[row][32 + hi * 16 + 8 + e];
    }
  }

  v8f o0 = {}; v8f o1 = {};

  for (int ch = 0; ch < NCH; ++ch) {
    const int cur = ch & 1, nxt = cur ^ 1;
    const bool more = (ch + 1) < NCH;
    // --- issue next chunk's global reads early (latency hidden by compute) ---
    if (more) {
      kvreg = *(const v16h*)(skp + (ch + 1) * TK);
      vvreg = *(const v16h*)(svp + (ch + 1) * TK);
      if (ch + 2 < NCH) {             // warm L2 for the chunk after
        __builtin_prefetch(skp + (ch + 2) * TK, 0, 0);
        __builtin_prefetch(svp + (ch + 2) * TK, 0, 0);
      }
    }

    // --- S = Q K^T : 2 output tiles per wave, K-dim = C = 2 x k32 ---
    v8f c0 = {}, c1 = {};
    {
      const v16h b00 = *(const v16h*)&KsB[cur][j0    ][0][hi][rr][0];
      const v16h b10 = *(const v16h*)&KsB[cur][j0 + 1][0][hi][rr][0];
      c0 = __builtin_amdgcn_wmma_f32_16x16x32_f16(false, aq0, false, b00, (short)0, c0, false, false);
      c1 = __builtin_amdgcn_wmma_f32_16x16x32_f16(false, aq0, false, b10, (short)0, c1, false, false);
      const v16h b01 = *(const v16h*)&KsB[cur][j0    ][1][hi][rr][0];
      const v16h b11 = *(const v16h*)&KsB[cur][j0 + 1][1][hi][rr][0];
      c0 = __builtin_amdgcn_wmma_f32_16x16x32_f16(false, aq1, false, b01, (short)0, c0, false, false);
      c1 = __builtin_amdgcn_wmma_f32_16x16x32_f16(false, aq1, false, b11, (short)0, c1, false, false);
    }
#pragma unroll
    for (int v = 0; v < 8; ++v) {     // C-frag: row = v + 8*hi
      const int row = ti * 16 + hi * 8 + v;
      Ss[row][j0 * 16 + rr]      = c0[v];
      Ss[row][j0 * 16 + 16 + rr] = c1[v];
    }
    __syncthreads();

    // --- online softmax: row chunk into registers, partial max ---
    const int r = tid & 63, q4 = tid >> 6;
    float sreg[16];
    {
      float m = -1e30f;
#pragma unroll
      for (int u = 0; u < 16; ++u) { sreg[u] = Ss[r][q4 * 16 + u]; m = fmaxf(m, sreg[u]); }
      pmax[q4][r] = m;
    }
    __syncthreads();
    if (tid < TQ) {
      const float cm   = fmaxf(fmaxf(pmax[0][tid], pmax[1][tid]),
                               fmaxf(pmax[2][tid], pmax[3][tid]));
      const float mnew = fmaxf(rowM[tid], cm);
      rowScale[tid] = __expf(rowM[tid] - mnew);
      rowM[tid]     = mnew;
    }
    __syncthreads();
    // --- P = exp(S - m) from registers, partial row sums, f16 into P tile ---
    {
      const float mrow = rowM[r];
      float ssum = 0.f;
#pragma unroll
      for (int u = 0; u < 16; ++u) {
        const float e = __expf(sreg[u] - mrow);
        QPs[r][q4 * 16 + u] = (_Float16)e;
        ssum += e;
      }
      psum[q4][r] = ssum;
    }
    __syncthreads();
    if (tid < TQ)
      rowL[tid] = rowL[tid] * rowScale[tid] +
                  psum[0][tid] + psum[1][tid] + psum[2][tid] + psum[3][tid];
    // --- rescale running O accumulator ---
#pragma unroll
    for (int v = 0; v < 8; ++v) {
      const float sc = rowScale[ti * 16 + hi * 8 + v];
      o0[v] *= sc; o1[v] *= sc;
    }
    // --- O += P V^T ---
#pragma unroll
    for (int s = 0; s < 2; ++s) {
      v16h a;
      const int row = ti * 16 + rr;
      const int cb  = s * 32 + hi * 16;
#pragma unroll
      for (int e = 0; e < 8; ++e) {
        a[e]     = QPs[row][cb + e];
        a[8 + e] = QPs[row][cb + 8 + e];
      }
      const v16h b0 = *(const v16h*)&VsB[cur][j0    ][s][hi][rr][0];
      const v16h b1 = *(const v16h*)&VsB[cur][j0 + 1][s][hi][rr][0];
      o0 = __builtin_amdgcn_wmma_f32_16x16x32_f16(false, a, false, b0, (short)0, o0, false, false);
      o1 = __builtin_amdgcn_wmma_f32_16x16x32_f16(false, a, false, b1, (short)0, o1, false, false);
    }
    // --- commit next chunk to the alternate buffers (no reader conflicts) ---
    if (more) {
#pragma unroll
      for (int u = 0; u < 16; ++u) KsB[nxt][kj][ks][kh][u][ke] = kvreg[u];
      *(v16h*)&VsB[nxt][vj][vs][vh][vn][0] = vvreg;
    }
    __syncthreads();   // publish next buffers; protect Ss / P tile
  }

  // --- epilogue: stage O in LDS, coalesced gamma*O/l + x store ---
#pragma unroll
  for (int v = 0; v < 8; ++v) {
    const int row = ti * 16 + hi * 8 + v;
    Ss[row][j0 * 16 + rr]      = o0[v];
    Ss[row][j0 * 16 + 16 + rr] = o1[v];
  }
  if (tid < TQ) rowScale[tid] = 1.0f / rowL[tid];
  __syncthreads();

  const float g = gamma[0];
  const int n = tid & 63;
#pragma unroll
  for (int cc = 0; cc < 16; ++cc) {
    const int c = (tid >> 6) * 16 + cc;
    const long ob = ((long)b * CDIM + c) * NPIX + qbase + n;
    outg[ob] = g * (Ss[n][c] * rowScale[n]) + xg[ob];
  }
}

// ---------------------------------------------------------------------------
extern "C" void kernel_launch(void* const* d_in, const int* in_sizes, int n_in,
                              void* d_out, int out_size, void* d_ws, size_t ws_size,
                              hipStream_t stream) {
  const float* x     = (const float*)d_in[0];
  const float* Wq    = (const float*)d_in[1];
  const float* bq    = (const float*)d_in[2];
  const float* Wk    = (const float*)d_in[3];
  const float* bk    = (const float*)d_in[4];
  const float* Wv    = (const float*)d_in[5];
  const float* bv    = (const float*)d_in[6];
  const float* gamma = (const float*)d_in[7];
  float* out = (float*)d_out;

  const size_t qBytes  = (size_t)2 * NPIX * CDIM * sizeof(_Float16);   // 4 MB
  const size_t kvBytes = (size_t)2 * CDIM * MKEYS * sizeof(_Float16);  // 1 MB
  _Float16* q = (_Float16*)d_ws;
  _Float16* k = (_Float16*)((char*)d_ws + qBytes);
  _Float16* v = (_Float16*)((char*)d_ws + qBytes + kvBytes);

  qkv_kernel<<<dim3((2 * NPIX) / 256), 256, 0, stream>>>(x, Wq, bq, q, 0);
  qkv_kernel<<<dim3((2 * MKEYS) / 256), 256, 0, stream>>>(x, Wk, bk, k, 1);
  qkv_kernel<<<dim3((2 * MKEYS) / 256), 256, 0, stream>>>(x, Wv, bv, v, 1);

  attn_kernel<<<dim3(NPIX / TQ, 2), 256, 0, stream>>>(q, k, v, x, gamma, out);
}
